// Net_21234318311808
// MI455X (gfx1250) — matrix operand order repaired and checked by reading, not compiled, not run on hardware
//
#include <hip/hip_runtime.h>

typedef __attribute__((ext_vector_type(16))) _Float16 v16h;
typedef __attribute__((ext_vector_type(8)))  _Float16 v8h;
typedef __attribute__((ext_vector_type(8)))  float    v8f;

// ---------------- degree / normalization ----------------
__global__ __launch_bounds__(256) void k_init_deg(float* deg, int n) {
    int i = blockIdx.x * 256 + threadIdx.x;
    if (i < n) deg[i] = 1.0f;               // self-loop contribution
}

__global__ __launch_bounds__(256) void k_accum_deg(const long long* __restrict__ ei,
                                                   float* deg, int E) {
    int e = blockIdx.x * 256 + threadIdx.x;
    if (e < E) atomicAdd(&deg[(long)ei[(long)E + e]], 1.0f);
}

__global__ __launch_bounds__(256) void k_dinv(float* deg, int n) {
    int i = blockIdx.x * 256 + threadIdx.x;
    if (i < n) deg[i] = rsqrtf(deg[i]);
}

// ---------------- WMMA GEMM: C[M,N] = A[M,K] * B[K,N] ----------------
// B (f32, row-major, small) is converted to f16 and transposed into LDS once
// per workgroup. Each wave computes a 16-row strip across all N columns.
// A fragment layout (16-bit A 16x32, ISA 7.12.2):
//   lane l: row = l&15, kg = l>>4;  VGPR0-3 = K kb..kb+7, VGPR4-7 = K kb+16..kb+23
//   with kb = kstep*32 + kg*8.
// B fragment: lane l: col = l&15; VGPR0-3 = K kb2..kb2+7, VGPR4-7 = K kb2+16..kb2+23,
//   kb2 = kstep*32 + kg*8  -> two 8-half runs in the transposed LDS image.
template<int K, int N, bool AHALF>
__global__ __launch_bounds__(256)
void k_gemm_wmma(const void* __restrict__ Aptr, const float* __restrict__ B,
                 float* __restrict__ C, int M)
{
    constexpr int NT     = N / 16;   // output n-tiles per wave
    constexpr int KSTEPS = K / 32;
    constexpr int KP     = K + 8;    // padded LDS K-stride (halves)
    __shared__ _Float16 Bt[N * KP];

    // cooperative load + transpose + f32->f16 convert of B into LDS
    for (int i = threadIdx.x; i < K * N; i += 256) {
        int n = i % N, k = i / N;                 // coalesced over n
        Bt[n * KP + k] = (_Float16)B[(long)k * N + n];
    }
    __syncthreads();

    const int wave = threadIdx.x >> 5;
    const int lane = threadIdx.x & 31;
    const int r    = lane & 15;
    const int kg   = lane >> 4;

    const long strip   = (long)blockIdx.x * 8 + wave;
    const bool valid   = (strip * 16) < (long)M;
    const long rowBase = valid ? strip * 16 : 0;
    long row = rowBase + r;
    if (row >= M) row = M - 1;                    // clamp (loads only)

    v8f acc[NT];
#pragma unroll
    for (int t = 0; t < NT; ++t)
#pragma unroll
        for (int j = 0; j < 8; ++j) acc[t][j] = 0.0f;

#pragma unroll
    for (int ks = 0; ks < KSTEPS; ++ks) {
        const int kb = ks * 32 + kg * 8;
        v16h a;
        if constexpr (AHALF) {
            const _Float16* A = (const _Float16*)Aptr + row * (long)K;
            v8h lo = *(const v8h*)(A + kb);
            v8h hi = *(const v8h*)(A + kb + 16);
#pragma unroll
            for (int j = 0; j < 8; ++j) { a[j] = lo[j]; a[8 + j] = hi[j]; }
        } else {
            const float* A = (const float*)Aptr + row * (long)K;
            const float4 f0 = *(const float4*)(A + kb);
            const float4 f1 = *(const float4*)(A + kb + 4);
            const float4 f2 = *(const float4*)(A + kb + 16);
            const float4 f3 = *(const float4*)(A + kb + 20);
            a[0]=(_Float16)f0.x;  a[1]=(_Float16)f0.y;  a[2]=(_Float16)f0.z;  a[3]=(_Float16)f0.w;
            a[4]=(_Float16)f1.x;  a[5]=(_Float16)f1.y;  a[6]=(_Float16)f1.z;  a[7]=(_Float16)f1.w;
            a[8]=(_Float16)f2.x;  a[9]=(_Float16)f2.y;  a[10]=(_Float16)f2.z; a[11]=(_Float16)f2.w;
            a[12]=(_Float16)f3.x; a[13]=(_Float16)f3.y; a[14]=(_Float16)f3.z; a[15]=(_Float16)f3.w;
        }
        const int kb2 = ks * 32 + kg * 8;
#pragma unroll
        for (int t = 0; t < NT; ++t) {
            const _Float16* bp = &Bt[(t * 16 + r) * KP + kb2];
            v8h b0 = *(const v8h*)bp;
            v8h b1 = *(const v8h*)(bp + 16);
            v16h b;
#pragma unroll
            for (int j = 0; j < 8; ++j) { b[j] = b0[j]; b[8 + j] = b1[j]; }
            acc[t] = __builtin_amdgcn_wmma_f32_16x16x32_f16(
                false, a, false, b, (short)0, acc[t], false, false);
        }
    }

    if (valid) {
#pragma unroll
        for (int t = 0; t < NT; ++t) {
            const long col = t * 16 + r;
#pragma unroll
            for (int j = 0; j < 8; ++j) {
                const long orow = rowBase + kg * 8 + j;   // C/D layout: lanes 16-31 -> M+8
                C[orow * (long)N + col] = acc[t][j];
            }
        }
    }
}

// ---------------- edge scatter: out[dst] += dinv[s]*dinv[d]*feat[src] ----------------
template<int C>
__global__ __launch_bounds__(256)
void k_scatter(const float* __restrict__ feat, const long long* __restrict__ ei,
               const float* __restrict__ dinv, float* __restrict__ out, int E)
{
    const int e = (int)(((long)blockIdx.x * 256 + threadIdx.x) >> 5);  // wave per edge
    if (e >= E) return;
    const int lane = threadIdx.x & 31;
    const long s = (long)ei[e];
    const long d = (long)ei[(long)E + e];
    const float nrm = dinv[s] * dinv[d];
    const float* f = feat + s * (long)C;
    float* o = out + d * (long)C;
#pragma unroll
    for (int j = 0; j < C / 32; ++j) {
        const int c = lane + j * 32;      // 128B coalesced per iteration
        atomicAdd(&o[c], nrm * f[c]);
    }
}

// ---------------- fused self-loop + bias (+relu), h stored as f16 ----------------
__global__ __launch_bounds__(256)
void k_fuse_relu_h16(const float* __restrict__ agg, const float* __restrict__ xw,
                     const float* __restrict__ dinv, const float* __restrict__ bias,
                     _Float16* __restrict__ h, long total)
{
    long i = (long)blockIdx.x * 256 + threadIdx.x;
    if (i >= total) return;
    const int  c    = (int)(i & 127);
    const long node = i >> 7;
    const float dv  = dinv[node];
    const float v   = agg[i] + dv * dv * xw[i] + bias[c];
    h[i] = (_Float16)fmaxf(v, 0.0f);
}

__global__ __launch_bounds__(256)
void k_fuse_z(const float* __restrict__ agg, const float* __restrict__ hw,
              const float* __restrict__ dinv, const float* __restrict__ bias,
              float* __restrict__ z, long total)
{
    long i = (long)blockIdx.x * 256 + threadIdx.x;
    if (i >= total) return;
    const int  c    = (int)(i & 63);
    const long node = i >> 6;
    const float dv  = dinv[node];
    z[i] = agg[i] + dv * dv * hw[i] + bias[c];
}

// ---------------- link scores: out[e] = dot(z[src], z[dst]) over 64 ch ----------------
__global__ __launch_bounds__(256)
void k_score(const float* __restrict__ z, const long long* __restrict__ eli,
             float* __restrict__ out, int E)
{
    const int e = (int)(((long)blockIdx.x * 256 + threadIdx.x) >> 5);   // wave per edge
    if (e >= E) return;
    const int lane = threadIdx.x & 31;
    const float* zs = z + (long)eli[e] * 64;
    const float* zd = z + (long)eli[(long)E + e] * 64;
    float acc = zs[lane] * zd[lane] + zs[lane + 32] * zd[lane + 32];
#pragma unroll
    for (int off = 16; off > 0; off >>= 1)
        acc += __shfl_xor(acc, off, 32);
    if (lane == 0) out[e] = acc;
}

extern "C" void kernel_launch(void* const* d_in, const int* in_sizes, int n_in,
                              void* d_out, int out_size, void* d_ws, size_t ws_size,
                              hipStream_t stream)
{
    (void)n_in; (void)out_size; (void)ws_size;
    const float*     x   = (const float*)d_in[0];
    const long long* ei  = (const long long*)d_in[1];
    const long long* eli = (const long long*)d_in[2];
    const float*     W1  = (const float*)d_in[3];
    const float*     b1  = (const float*)d_in[4];
    const float*     W2  = (const float*)d_in[5];
    const float*     b2  = (const float*)d_in[6];

    const int N  = in_sizes[0] / 256;   // 100000 nodes (in_c = 256)
    const int E  = in_sizes[1] / 2;     // 1,600,000 edges
    const int EL = in_sizes[2] / 2;     // 200,000 label edges

    // workspace layout (L2-resident intermediates)
    char* ws = (char*)d_ws;
    float*    dinv = (float*)ws;                                          // N*4
    float*    bufA = (float*)(ws + (size_t)512 * 1024);                   // xw / hw : N*128*4
    float*    bufB = (float*)(ws + (size_t)512 * 1024 + (size_t)N*128*4); // agg     : N*128*4
    _Float16* h16  = (_Float16*)(ws + (size_t)512*1024 + (size_t)N*128*8);// h f16 -> later z f32
    float*    z    = (float*)h16;

    // 1) symmetric normalization coefficients
    k_init_deg <<<(N + 255) / 256, 256, 0, stream>>>(dinv, N);
    k_accum_deg<<<(E + 255) / 256, 256, 0, stream>>>(ei, dinv, E);
    k_dinv     <<<(N + 255) / 256, 256, 0, stream>>>(dinv, N);

    const int strips      = (N + 15) / 16;
    const int gemm_blocks = (strips + 7) / 8;

    // 2) layer 1: xw = x @ W1 (WMMA f16 -> f32)
    k_gemm_wmma<256, 128, false><<<gemm_blocks, 256, 0, stream>>>(x, W1, bufA, N);
    hipMemsetAsync(bufB, 0, (size_t)N * 128 * 4, stream);
    k_scatter<128><<<(E + 7) / 8, 256, 0, stream>>>(bufA, ei, dinv, bufB, E);
    k_fuse_relu_h16<<<(int)(((long)N * 128 + 255) / 256), 256, 0, stream>>>(
        bufB, bufA, dinv, b1, h16, (long)N * 128);

    // 3) layer 2: hw = h @ W2 (A already f16)
    k_gemm_wmma<128, 64, true><<<gemm_blocks, 256, 0, stream>>>(h16, W2, bufA, N);
    hipMemsetAsync(bufB, 0, (size_t)N * 64 * 4, stream);
    k_scatter<64><<<(E + 7) / 8, 256, 0, stream>>>(bufA, ei, dinv, bufB, E);
    k_fuse_z<<<(int)(((long)N * 64 + 255) / 256), 256, 0, stream>>>(
        bufB, bufA, dinv, b2, z, (long)N * 64);

    // 4) link prediction scores
    k_score<<<(EL + 7) / 8, 256, 0, stream>>>(z, eli, (float*)d_out, EL);
}
